// GCPN_crem_86122684220332
// MI455X (gfx1250) — matrix operand order repaired and checked by reading, not compiled, not run on hardware
//
#include <hip/hip_runtime.h>
#include <math.h>

#define HEADS 4
#define CH 128
#define HC 512           // HEADS*CH
#define ATT_STRIDE 257   // 2*CH+1
#define EPS_BN 1e-5f
#define NEG_SLOPE 0.2f

typedef __attribute__((ext_vector_type(16))) _Float16 v16h;
typedef __attribute__((ext_vector_type(8)))  float    v8f;

// ---------- utility ----------
__global__ void k_zero_f32(float* __restrict__ p, long n) {
  long i = (long)blockIdx.x * blockDim.x + threadIdx.x;
  if (i < n) p[i] = 0.f;
}

__device__ __forceinline__ unsigned fkey(float f) {
  unsigned u = __float_as_uint(f);
  return (u & 0x80000000u) ? ~u : (u | 0x80000000u);
}
__device__ __forceinline__ float fdec(unsigned k) {
  unsigned u = (k & 0x80000000u) ? (k & 0x7FFFFFFFu) : ~k;
  return __uint_as_float(u);
}

// ---------- pack W[K,N] (f32, row-major) into f16 WMMA-fragment order ----------
// Layout: for col-tile tn (16 cols), k-tile tk (32 ks), lane l (0..31):
//   16 contiguous halves; element i -> k = tk*32 + (l>>4)*8 + (i&7) + (i>>3)*16, col = tn*16 + (l&15)
__global__ void k_pack_w(const float* __restrict__ W, _Float16* __restrict__ Wp, int K, int N) {
  long idx = (long)blockIdx.x * blockDim.x + threadIdx.x;
  if (idx >= (long)K * N) return;
  int i = (int)(idx & 15);
  int l = (int)((idx >> 4) & 31);
  long rest = idx >> 9;
  int ktiles = K >> 5;
  int tk = (int)(rest % ktiles);
  int tn = (int)(rest / ktiles);
  int k   = tk * 32 + (l >> 4) * 8 + (i & 7) + (i >> 3) * 16;
  int col = tn * 16 + (l & 15);
  Wp[idx] = (_Float16)W[(size_t)k * N + col];
}

// ---------- WMMA GEMM: C = act( (BN? A*scale+shift : A)[M,K] @ W[K,N] + bias ) ----------
// block (32,8): 8 waves; each wave owns TWO 16x16 N-tiles (reuses its A fragment).
// Block covers 16 rows x 256 cols. K mult of 32, N mult of 256, M guarded (branch-free loads).
template <int BN>
__global__ void k_gemm(const float* __restrict__ A, const _Float16* __restrict__ Wp,
                       const float* __restrict__ bnScale, const float* __restrict__ bnShift,
                       const float* __restrict__ bias, float* __restrict__ C,
                       int M, int K, int N, int doRelu)
{
  const int lane = threadIdx.x;
  const int wid  = threadIdx.y;
  const int l16  = lane & 15;
  const int half = lane >> 4;
  const int rowBase = blockIdx.x * 16;
  const int colBase = blockIdx.y * 256 + wid * 16;   // tile0 cols; tile1 = +128
  const int aRow = rowBase + l16;
  const bool rowOK = aRow < M;
  const int safeRow = rowOK ? aRow : 0;
  const float msk = rowOK ? 1.f : 0.f;
  const int ktiles = K >> 5;
  const float4* __restrict__ Arow = (const float4*)(A + (size_t)safeRow * K);
  const _Float16* __restrict__ wp0 = Wp + ((((size_t)(colBase >> 4)) * ktiles) << 9) + (lane << 4);
  const _Float16* __restrict__ wp1 = wp0 + (((size_t)8 * ktiles) << 9);  // +8 col-tiles

  v8f acc0 = {}, acc1 = {};
  for (int tk = 0; tk < ktiles; ++tk) {
    const int kA = tk * 32 + half * 8;
    float fa[16];
    *(float4*)(fa + 0)  = Arow[(kA >> 2) + 0];
    *(float4*)(fa + 4)  = Arow[(kA >> 2) + 1];
    *(float4*)(fa + 8)  = Arow[(kA >> 2) + 4];      // kA+16
    *(float4*)(fa + 12) = Arow[(kA >> 2) + 5];
    v16h a;
#pragma unroll
    for (int i = 0; i < 8; ++i) {
      float f0 = fa[i]     * msk;
      float f1 = fa[8 + i] * msk;
      if (BN) {
        f0 = f0 * bnScale[kA + i]      + bnShift[kA + i];
        f1 = f1 * bnScale[kA + 16 + i] + bnShift[kA + 16 + i];
      }
      a[i]     = (_Float16)f0;
      a[i + 8] = (_Float16)f1;
    }
    v16h b0 = *(const v16h*)(wp0 + ((size_t)tk << 9));
    v16h b1 = *(const v16h*)(wp1 + ((size_t)tk << 9));
    acc0 = __builtin_amdgcn_wmma_f32_16x16x32_f16(false, a, false, b0, (short)0, acc0, false, false);
    acc1 = __builtin_amdgcn_wmma_f32_16x16x32_f16(false, a, false, b1, (short)0, acc1, false, false);
  }
  const int c0 = colBase + l16, c1 = c0 + 128;
  const float bv0 = bias ? bias[c0] : 0.f;
  const float bv1 = bias ? bias[c1] : 0.f;
#pragma unroll
  for (int r = 0; r < 8; ++r) {
    int row = rowBase + half * 8 + r;   // C layout: VGPR r -> row r (+8 for lanes 16..31)
    if (row < M) {
      float v0 = acc0[r] + bv0, v1 = acc1[r] + bv1;
      if (doRelu) { v0 = fmaxf(v0, 0.f); v1 = fmaxf(v1, 0.f); }
      C[(size_t)row * N + c0] = v0;
      C[(size_t)row * N + c1] = v1;
    }
  }
}

// ---------- BatchNorm stats -> fused scale/shift ----------
__global__ void k_bn_stats(const float* __restrict__ h, const float* __restrict__ w,
                           const float* __restrict__ b, float* __restrict__ scale,
                           float* __restrict__ shift, int n)
{
  const int c = blockIdx.x;
  const int t = threadIdx.x;
  float s = 0.f, ss = 0.f;
  for (int r = t; r < n; r += blockDim.x) {
    float v = h[(size_t)r * CH + c];
    s += v; ss += v * v;
  }
  __shared__ float s1[256], s2[256];
  s1[t] = s; s2[t] = ss; __syncthreads();
  for (int o = 128; o; o >>= 1) {
    if (t < o) { s1[t] += s1[t + o]; s2[t] += s2[t + o]; }
    __syncthreads();
  }
  if (t == 0) {
    float mu  = s1[0] / n;
    float var = s2[0] / n - mu * mu;
    float inv = rsqrtf(var + EPS_BN);
    float sc  = w[c] * inv;
    scale[c] = sc;
    shift[c] = b[c] - mu * sc;
  }
}

// ---------- attention scores per (node, head) ----------
__global__ void k_scores(const float* __restrict__ xh, const float* __restrict__ att,
                         float* __restrict__ s_src, float* __restrict__ s_dst, int n)
{
  const int node = blockIdx.x;
  const int h    = threadIdx.x >> 5;
  const int lane = threadIdx.x & 31;
  const float* base = xh + ((size_t)node * HEADS + h) * CH;
  const float* as = att + (size_t)h * ATT_STRIDE;
  const float* ad = as + CH;
  float ssrc = 0.f, sdst = 0.f;
  for (int c = lane; c < CH; c += 32) {
    float v = base[c];
    ssrc += v * as[c];
    sdst += v * ad[c];
  }
  for (int o = 16; o; o >>= 1) {
    ssrc += __shfl_xor(ssrc, o);
    sdst += __shfl_xor(sdst, o);
  }
  if (lane == 0) {
    s_src[node * HEADS + h] = ssrc;
    s_dst[node * HEADS + h] = sdst;
  }
}

// ---------- edge logits + leaky relu + segment max ----------
__global__ void k_edge_logits_max(const int* __restrict__ src, const int* __restrict__ dst,
                                  const float* __restrict__ eattr, const float* __restrict__ linE,
                                  const float* __restrict__ att, const float* __restrict__ s_src,
                                  const float* __restrict__ s_dst, float* __restrict__ logitbuf,
                                  unsigned* __restrict__ mkey, int E)
{
  long idx = (long)blockIdx.x * blockDim.x + threadIdx.x;
  if (idx >= (long)E * HEADS) return;
  int e = (int)(idx >> 2), h = (int)(idx & 3);
  float ae = att[h * ATT_STRIDE + 2 * CH];
  float lg = s_src[src[e] * HEADS + h] + s_dst[dst[e] * HEADS + h] + eattr[e] * linE[h] * ae;
  lg = (lg > 0.f) ? lg : NEG_SLOPE * lg;
  logitbuf[idx] = lg;
  atomicMax(&mkey[dst[e] * HEADS + h], fkey(lg));
}

// ---------- exp(logit - max) + segment sum ----------
__global__ void k_edge_exp(const int* __restrict__ dst, float* __restrict__ logitbuf,
                           const unsigned* __restrict__ mkey, float* __restrict__ ssum, int E)
{
  long idx = (long)blockIdx.x * blockDim.x + threadIdx.x;
  if (idx >= (long)E * HEADS) return;
  int e = (int)(idx >> 2), h = (int)(idx & 3);
  float m = fdec(mkey[dst[e] * HEADS + h]);
  if (!isfinite(m)) m = 0.f;
  float ex = __expf(logitbuf[idx] - m);
  logitbuf[idx] = ex;
  atomicAdd(&ssum[dst[e] * HEADS + h], ex);
}

// ---------- scatter: agg[dst] += alpha * xh[src], float4 granularity ----------
__global__ void k_scatter_msg(const int* __restrict__ src, const int* __restrict__ dst,
                              const float* __restrict__ exbuf, const float* __restrict__ ssum,
                              const float* __restrict__ xh, float* __restrict__ agg, long total)
{
  long t = (long)blockIdx.x * blockDim.x + threadIdx.x;
  if (t >= total) return;                 // total = E * 128 quads
  int q = (int)(t & 127);                 // quad index within 512 channels
  int e = (int)(t >> 7);
  int h = q >> 5;                         // 32 quads per head
  int d = dst[e];
  float alpha = exbuf[e * HEADS + h] / (ssum[d * HEADS + h] + 1e-16f);
  const float4 v = *(const float4*)(xh + (size_t)src[e] * HC + (q << 2));
  float* ap = agg + (size_t)d * HC + (q << 2);
  atomicAdd(ap + 0, alpha * v.x);
  atomicAdd(ap + 1, alpha * v.y);
  atomicAdd(ap + 2, alpha * v.z);
  atomicAdd(ap + 3, alpha * v.w);
}

// ---------- layer epilogue: relu(mean_h(agg) + bias + mean_h(xh)) ----------
__global__ void k_epilogue(const float* __restrict__ agg, const float* __restrict__ xh,
                           const float* __restrict__ bias, float* __restrict__ out, int n)
{
  long idx = (long)blockIdx.x * blockDim.x + threadIdx.x;
  if (idx >= (long)n * CH) return;
  int nd = (int)(idx >> 7), c = (int)(idx & 127);
  float a = 0.f, xm = 0.f;
#pragma unroll
  for (int h = 0; h < HEADS; ++h) {
    a  += agg[(size_t)nd * HC + h * CH + c];
    xm += xh [(size_t)nd * HC + h * CH + c];
  }
  float v = a * 0.25f + bias[c] + xm * 0.25f;
  out[(size_t)nd * CH + c] = fmaxf(v, 0.f);
}

// ---------- pooling ----------
__global__ void k_pool_count(const int* __restrict__ batch, float* __restrict__ cnt, int n) {
  int i = blockIdx.x * blockDim.x + threadIdx.x;
  if (i < n) atomicAdd(&cnt[batch[i]], 1.f);
}
__global__ void k_pool_sum(const int* __restrict__ batch, const float* __restrict__ emb,
                           float* __restrict__ pooled, int n, int D) {
  long idx = (long)blockIdx.x * blockDim.x + threadIdx.x;
  if (idx >= (long)n * D) return;
  int nd = (int)(idx / D), c = (int)(idx % D);
  atomicAdd(&pooled[(size_t)batch[nd] * D + c], emb[idx]);
}
__global__ void k_pool_div(float* __restrict__ pooled, const float* __restrict__ cnt, int B, int D) {
  int idx = blockIdx.x * blockDim.x + threadIdx.x;
  if (idx >= B * D) return;
  pooled[idx] /= fmaxf(cnt[idx / D], 1.f);
}

// ---------- final 512->1 linear + softmax over graphs ----------
__global__ void k_final_softmax(const float* __restrict__ z, const float* __restrict__ wf,
                                const float* __restrict__ bf, float* __restrict__ out, int B, int K)
{
  __shared__ float red[512];
  const int t = threadIdx.x;
  float myl = 0.f;
  if (t < B) {
    float s = 0.f;
    for (int k = 0; k < K; ++k) s += z[(size_t)t * K + k] * wf[k];
    myl = s + bf[0];
  }
  red[t] = (t < B) ? myl : -INFINITY;
  __syncthreads();
  for (int o = 256; o; o >>= 1) { if (t < o) red[t] = fmaxf(red[t], red[t + o]); __syncthreads(); }
  float m = red[0];
  __syncthreads();
  float ex = (t < B) ? __expf(myl - m) : 0.f;
  red[t] = ex;
  __syncthreads();
  for (int o = 256; o; o >>= 1) { if (t < o) red[t] += red[t + o]; __syncthreads(); }
  if (t < B) out[t] = ex / red[0];
}

// ========================= driver =========================
extern "C" void kernel_launch(void* const* d_in, const int* in_sizes, int n_in,
                              void* d_out, int out_size, void* d_ws, size_t ws_size,
                              hipStream_t stream)
{
  const float* x      = (const float*)d_in[0];
  const float* eattr  = (const float*)d_in[1];
  const int*   eidx   = (const int*)d_in[2];
  const int*   batch  = (const int*)d_in[3];
  const float* linN[3] = {(const float*)d_in[4], (const float*)d_in[8],  (const float*)d_in[12]};
  const float* linE[3] = {(const float*)d_in[5], (const float*)d_in[9],  (const float*)d_in[13]};
  const float* att [3] = {(const float*)d_in[6], (const float*)d_in[10], (const float*)d_in[14]};
  const float* bias[3] = {(const float*)d_in[7], (const float*)d_in[11], (const float*)d_in[15]};
  const float* bn1w = (const float*)d_in[16]; const float* bn1b = (const float*)d_in[17];
  const float* bn2w = (const float*)d_in[18]; const float* bn2b = (const float*)d_in[19];
  const float* fw = (const float*)d_in[20];   const float* fb = (const float*)d_in[21];
  const float* w0 = (const float*)d_in[22];   const float* b0 = (const float*)d_in[23];
  const float* w1 = (const float*)d_in[24];   const float* b1 = (const float*)d_in[25];
  const float* w2 = (const float*)d_in[26];   const float* b2 = (const float*)d_in[27];
  const float* wf = (const float*)d_in[28];   const float* bf = (const float*)d_in[29];

  const int N = in_sizes[0] / CH;     // 20000 nodes
  const int E = in_sizes[1];          // 340000 edges
  const int B = out_size;             // 500 graphs
  const int* srcI = eidx;
  const int* dstI = eidx + E;
  float* outp = (float*)d_out;

  // workspace layout (floats)
  float* ws = (float*)d_ws;
  size_t o = 0;
  float*    xh      = ws + o; o += (size_t)N * HC;
  float*    agg     = ws + o; o += (size_t)N * HC;
  float*    hA      = ws + o; o += (size_t)N * CH;
  float*    hB      = ws + o; o += (size_t)N * CH;
  float*    ssrc    = ws + o; o += (size_t)N * HEADS;
  float*    sdst    = ws + o; o += (size_t)N * HEADS;
  unsigned* mkey    = (unsigned*)(ws + o); o += (size_t)N * HEADS;
  float*    ssum    = ws + o; o += (size_t)N * HEADS;
  float*    exbuf   = ws + o; o += (size_t)E * HEADS;
  float*    bnScale = ws + o; o += CH;
  float*    bnShift = ws + o; o += CH;
  float*    pooled  = ws + o; o += (size_t)B * 256;
  float*    cnt     = ws + o; o += B;
  o = (o + 15) & ~(size_t)15;         // align pack buffer to 64B
  _Float16* wpk     = (_Float16*)(ws + o); o += (512 * 512) / 2;  // f16 pack scratch (max K*N)
  float*    z0      = ws + o; o += (size_t)B * 512;
  float*    z1      = ws + o; o += (size_t)B * 512;
  float*    emb     = xh;             // reuse after layers finish

  auto zblk = [](long n) { return (unsigned)((n + 255) / 256); };
  auto zero = [&](float* p, long n) { k_zero_f32<<<zblk(n), 256, 0, stream>>>(p, n); };
  auto pack = [&](const float* W, int K, int Nn) {
    k_pack_w<<<zblk((long)K * Nn), 256, 0, stream>>>(W, wpk, K, Nn);
  };

  auto run_layer = [&](const float* xin, int l, const float* bnw, const float* bnb, float* hout) {
    pack(linN[l], CH, HC);
    if (bnw) {
      k_bn_stats<<<CH, 256, 0, stream>>>(xin, bnw, bnb, bnScale, bnShift, N);
      k_gemm<1><<<dim3(N / 16, HC / 256), dim3(32, 8), 0, stream>>>(
          xin, wpk, bnScale, bnShift, nullptr, xh, N, CH, HC, 0);
    } else {
      k_gemm<0><<<dim3(N / 16, HC / 256), dim3(32, 8), 0, stream>>>(
          xin, wpk, nullptr, nullptr, nullptr, xh, N, CH, HC, 0);
    }
    k_scores<<<N, 128, 0, stream>>>(xh, att[l], ssrc, sdst, N);
    zero(agg, (long)N * HC);
    zero((float*)mkey, (long)N * HEADS);
    zero(ssum, (long)N * HEADS);
    k_edge_logits_max<<<zblk((long)E * HEADS), 256, 0, stream>>>(
        srcI, dstI, eattr, linE[l], att[l], ssrc, sdst, exbuf, mkey, E);
    k_edge_exp<<<zblk((long)E * HEADS), 256, 0, stream>>>(dstI, exbuf, mkey, ssum, E);
    k_scatter_msg<<<zblk((long)E * CH), 256, 0, stream>>>(
        srcI, dstI, exbuf, ssum, xh, agg, (long)E * CH);
    k_epilogue<<<zblk((long)N * CH), 256, 0, stream>>>(agg, xh, bias[l], hout, N);
  };

  run_layer(x,  0, nullptr, nullptr, hA);
  run_layer(hA, 1, bn1w, bn1b, hB);
  run_layer(hB, 2, bn2w, bn2b, hA);

  // emb = hA @ fw + fb   (20000x128 @ 128x256)
  pack(fw, CH, 256);
  k_gemm<0><<<dim3(N / 16, 256 / 256), dim3(32, 8), 0, stream>>>(
      hA, wpk, nullptr, nullptr, fb, emb, N, CH, 256, 0);

  // mean pool per graph
  zero(pooled, (long)B * 256);
  zero(cnt, B);
  k_pool_count<<<zblk(N), 256, 0, stream>>>(batch, cnt, N);
  k_pool_sum<<<zblk((long)N * 256), 256, 0, stream>>>(batch, emb, pooled, N, 256);
  k_pool_div<<<zblk((long)B * 256), 256, 0, stream>>>(pooled, cnt, B, 256);

  // MLP: 256 -> 512 -> 512 -> 512 (relu), M = 500 guarded
  pack(w0, 256, 512);
  k_gemm<0><<<dim3((B + 15) / 16, 2), dim3(32, 8), 0, stream>>>(
      pooled, wpk, nullptr, nullptr, b0, z0, B, 256, 512, 1);
  pack(w1, 512, 512);
  k_gemm<0><<<dim3((B + 15) / 16, 2), dim3(32, 8), 0, stream>>>(
      z0, wpk, nullptr, nullptr, b1, z1, B, 512, 512, 1);
  pack(w2, 512, 512);
  k_gemm<0><<<dim3((B + 15) / 16, 2), dim3(32, 8), 0, stream>>>(
      z1, wpk, nullptr, nullptr, b2, z0, B, 512, 512, 1);

  // final linear + softmax over the 500 graphs
  k_final_softmax<<<1, 512, 0, stream>>>(z0, wf, bf, outp, B, 512);

  (void)n_in; (void)ws_size;
}
